// Generator2_28690381537388
// MI455X (gfx1250) — compile-verified, matrix operand order checked by reading, not run on hardware
//
#include <hip/hip_runtime.h>
#include <hip/hip_bf16.h>

// ---------------- types ----------------
typedef __attribute__((ext_vector_type(16))) __bf16 bf16x16;
typedef __attribute__((ext_vector_type(8)))  __bf16 bf16x8;
typedef __attribute__((ext_vector_type(8)))  float  f32x8;

#define K_USERS 256
#define NB      64
#define MBS     20
#define BPOS    (NB*MBS)      // 1280
#define H       256
#define H3      768
#define L       2
#define MTOT    (K_USERS*NB)  // 16384 rows for gi GEMM
#define NTILES  (H3/16)       // 48 column tiles
#define KCHUNKS (H/32)        // 8 k-chunks per WMMA reduction
#define WPL     ((size_t)H3*H) // packed elements per layer per matrix

// Load one WMMA A fragment (16x32 bf16) from row-major [m, k] memory.
// Lane layout per CDNA5 ISA 7.12.2:
//   lanes 0-15 : k+0..7 in v0-3, k+16..23 in v4-7
//   lanes 16-31: k+8..15 in v0-3, k+24..31 in v4-7
__device__ __forceinline__ bf16x16 load_afrag(const __bf16* p, int lane) {
    p += ((lane >> 4) & 1) * 8;
    bf16x16 f;
    *(bf16x8*)&f         = *(const bf16x8*)p;
    *(((bf16x8*)&f) + 1) = *(const bf16x8*)(p + 16);
    return f;
}

// Load one WMMA B fragment from the fragment-packed weight buffer:
// fragment (nt, kc) is 512 bf16 = 1KB contiguous; lane's 16 elements are
// 32 contiguous bytes at base + ((nt*8+kc)*32 + lane)*16.
__device__ __forceinline__ bf16x16 load_bfrag(const __bf16* base, int nt, int kc, int lane) {
    const __bf16* p = base + (((size_t)(nt * KCHUNKS + kc) * 32 + lane) << 4);
    bf16x16 f;
    *(bf16x8*)&f         = *(const bf16x8*)p;
    *(((bf16x8*)&f) + 1) = *(const bf16x8*)(p + 8);
    return f;
}

__device__ __forceinline__ f32x8 wmma_bf16(bf16x16 a, bf16x16 b, f32x8 c) {
    return __builtin_amdgcn_wmma_f32_16x16x32_bf16(
        /*neg_a=*/false, a, /*neg_b=*/false, b,
        /*c_mod=*/(short)0, c, /*reuse_a=*/false, /*reuse_b=*/false);
}

// ---------------- kernel 1: weight f32 -> fragment-packed bf16 ----------------
// w: [L, 3H, H] row-major ([n, k]).  wp: [L][nt][kc][lane][e] packed.
__global__ void k_pack(const float* __restrict__ w, __bf16* __restrict__ wp) {
    int i = blockIdx.x * 256 + threadIdx.x;       // over L * 3H * H
    int l = i / (int)WPL;
    int r = i - l * (int)WPL;
    int e    = r & 15;
    int lane = (r >> 4) & 31;
    int kc   = (r >> 9) & 7;
    int nt   = r >> 12;                            // 0..47
    int n = nt * 16 + (lane & 15);
    int k = kc * 32 + (lane >> 4) * 8 + e + ((e >= 8) ? 8 : 0);
    wp[i] = (__bf16)w[(size_t)l * WPL + (size_t)n * H + k];
}

// ---------------- kernel 2: basket pooling ----------------
__global__ void k_basket(const float* __restrict__ fbp, const int* __restrict__ seq,
                         const float* __restrict__ embed,
                         float* __restrict__ bke, float* __restrict__ fmax) {
    int blk = blockIdx.x;            // k*NB + nb
    int k  = blk >> 6;
    int nb = blk & 63;
    int h  = threadIdx.x;
    int base = k * BPOS + nb * MBS;
    float acc = 0.f, psum = 0.f, pmax = 0.f;
    #pragma unroll 4
    for (int j = 0; j < MBS; ++j) {
        int id  = seq[base + j];                 // -1 = padding
        float p = fbp[base + j];
        p = (id != -1) ? p : 0.f;
        acc  += p * embed[(long)(id + 1) * H + h];
        psum += p;
        pmax  = fmaxf(pmax, p);
    }
    bke[(long)blk * H + h] = acc / (psum + 1e-10f);
    if (h == 0) fmax[blk] = pmax;
}

// ---------------- kernel 3: stable valid-first order + lengths ----------------
__global__ void k_order(const float* __restrict__ fmax, int* __restrict__ order,
                        int* __restrict__ lengths) {
    int k = threadIdx.x;             // 256 threads, one per user
    int cnt = 0;
    for (int t = 0; t < NB; ++t)
        if (fmax[k * NB + t] > 0.f) order[k * NB + cnt++] = t;
    int len = cnt ? cnt : 1;
    for (int t = 0; t < NB; ++t)
        if (!(fmax[k * NB + t] > 0.f)) order[k * NB + cnt++] = t;
    lengths[k] = len;
}

// ---------------- kernel 4: reorder + convert to bf16 ----------------
__global__ void k_reorder(const float* __restrict__ bke, const int* __restrict__ order,
                          __bf16* __restrict__ xbf) {
    int i = blockIdx.x * 256 + threadIdx.x;  // over K*NB*H
    int h = i & (H - 1);
    int t = (i >> 8) & (NB - 1);
    int k = i >> 14;
    int src = order[k * NB + t];
    xbf[i] = (__bf16)bke[((long)(k * NB + src)) * H + h];
}

// ---------------- kernel 5: gi = x @ W_ih^T + b_ih  (WMMA GEMM) ----------------
// M = 16384, N = 768, Kd = 256. 8 waves as 2(M) x 4(N); wave tile 16x64.
// B-fragment stream prefetched 2 k-chunks ahead into the WGP cache.
__global__ void __launch_bounds__(256)
__attribute__((amdgpu_waves_per_eu(4)))
k_gi(const __bf16* __restrict__ x,
     const __bf16* __restrict__ wpack,
     const float* __restrict__ bias,
     float* __restrict__ gi) {
    int lane = threadIdx.x & 31;
    int wave = threadIdx.x >> 5;
    int wy = wave >> 2, wx = wave & 3;
    int m0 = blockIdx.x * 32 + wy * 16;
    int n0 = blockIdx.y * 256 + wx * 64;
    int ntb = n0 >> 4;
    int mr = m0 + (lane & 15);

    // preload all 8 A fragments (this wave's 16x256 row block)
    bf16x16 areg[KCHUNKS];
    #pragma unroll
    for (int kc = 0; kc < KCHUNKS; ++kc)
        areg[kc] = load_afrag(x + (size_t)mr * H + kc * 32, lane);

    f32x8 acc[4] = {};
    bf16x16 bcur[4], bnxt[4];
    #pragma unroll
    for (int j = 0; j < 4; ++j) bcur[j] = load_bfrag(wpack, ntb + j, 0, lane);

    #pragma unroll
    for (int kc = 0; kc < KCHUNKS; ++kc) {
        if (kc < KCHUNKS - 2) {
            // pull the k-chunk after next into WGP$ while WMMAs run
            #pragma unroll
            for (int j = 0; j < 4; ++j)
                __builtin_prefetch(
                    wpack + (((size_t)((ntb + j) * KCHUNKS + kc + 2) * 32 + lane) << 4), 0, 3);
        }
        if (kc < KCHUNKS - 1) {
            #pragma unroll
            for (int j = 0; j < 4; ++j) bnxt[j] = load_bfrag(wpack, ntb + j, kc + 1, lane);
        }
        #pragma unroll
        for (int j = 0; j < 4; ++j) acc[j] = wmma_bf16(areg[kc], bcur[j], acc[j]);
        #pragma unroll
        for (int j = 0; j < 4; ++j) bcur[j] = bnxt[j];
    }

    int rowbase = m0 + 8 * (lane >> 4);
    #pragma unroll
    for (int j = 0; j < 4; ++j) {
        int col = n0 + j * 16 + (lane & 15);
        float bv = bias[col];
        #pragma unroll
        for (int r = 0; r < 8; ++r)
            gi[(size_t)(rowbase + r) * H3 + col] = acc[j][r] + bv;
    }
}

// ---------------- kernel 6: recurrent GRU scan ----------------
// One block = 16 users, full T=64 scan. 512 threads = 16 waves.
// Wave w owns column tile w of ALL THREE gates (N-tiles w, w+16, w+32), so
// after its 3 WMMA accumulators finish, each lane holds the r/z/n
// pre-activations for exactly the (row, col) h-elements it updates itself:
//   - W_hh fragments resident in VGPRs for the whole 64-step scan
//   - h_prev resident in 8 VGPRs per lane (producer == consumer lane)
//   - only LDS traffic: the 8KB bf16 h tile (WMMA A operand)
//   - this step's gi gate rows prefetched before the WMMA phase so the
//     gate-phase loads (on the serial critical path) hit cache
__global__ void __launch_bounds__(512)
__attribute__((amdgpu_waves_per_eu(4)))
k_scan(const float* __restrict__ gi,
       const __bf16* __restrict__ whh_pack,
       const float* __restrict__ bhh,
       float* __restrict__ hout,
       __bf16* __restrict__ xbf) {
    __shared__ __bf16 hbf[16][H];    // 8 KB : h in bf16 (WMMA A source)

    int tid  = threadIdx.x;
    int lane = tid & 31;
    int wave = tid >> 5;             // 0..15
    int kg   = blockIdx.x * 16;      // user base
    int nrl  = lane & 15;
    int col  = wave * 16 + nrl;      // this lane's gate column (0..255)
    int rb   = 8 * (lane >> 4);      // C-fragment row base (0 or 8)

    // ---- preload this wave's 24 W_hh fragments (r, z, n tiles) ----
    bf16x16 breg[3][KCHUNKS];
    #pragma unroll
    for (int g = 0; g < 3; ++g)
        #pragma unroll
        for (int kc = 0; kc < KCHUNKS; ++kc)
            breg[g][kc] = load_bfrag(whh_pack, g * 16 + wave, kc, lane);

    // per-lane bias constants
    float bR = bhh[col], bZ = bhh[H + col], bN = bhh[2 * H + col];
    float hprev[8];
    #pragma unroll
    for (int r = 0; r < 8; ++r) hprev[r] = 0.f;

    for (int i = tid; i < 16 * H; i += 512) ((__bf16*)hbf)[i] = (__bf16)0.f;
    __syncthreads();

    for (int t = 0; t < NB; ++t) {
        // prefetch this step's gate inputs; they are consumed after the WMMAs
        #pragma unroll
        for (int r = 0; r < 8; ++r) {
            const float* gir = gi + ((size_t)(kg + rb + r) * NB + t) * H3;
            __builtin_prefetch(gir + col,         0, 0);
            __builtin_prefetch(gir + H + col,     0, 0);
            __builtin_prefetch(gir + 2 * H + col, 0, 0);
        }

        // ---- gh tiles for this wave's columns; weights already in VGPRs ----
        f32x8 aR = {}, aZ = {}, aN = {};
        #pragma unroll
        for (int kc = 0; kc < KCHUNKS; ++kc) {
            bf16x16 a = load_afrag(&hbf[nrl][kc * 32], lane);   // ds_load_b128 x2
            aR = wmma_bf16(a, breg[0][kc], aR);
            aZ = wmma_bf16(a, breg[1][kc], aZ);
            aN = wmma_bf16(a, breg[2][kc], aN);
        }
        __syncthreads();             // all h reads complete before overwrite

        // ---- gates: fully register-local except gi loads / h stores ----
        #pragma unroll
        for (int r = 0; r < 8; ++r) {
            int m = rb + r;
            size_t rowg = (size_t)(kg + m) * NB + t;
            const float* gir = gi + rowg * H3;
            float grv = gir[col]         + aR[r] + bR;
            float gzv = gir[H + col]     + aZ[r] + bZ;
            float hn  =                    aN[r] + bN;
            float rr = 1.f / (1.f + __expf(-grv));
            float zz = 1.f / (1.f + __expf(-gzv));
            float nn = tanhf(gir[2 * H + col] + rr * hn);
            float hnew = (1.f - zz) * nn + zz * hprev[r];
            hprev[r] = hnew;
            hbf[m][col] = (__bf16)hnew;                 // next step's A operand
            hout[rowg * H + col] = hnew;                // layer output (f32)
            xbf [rowg * H + col] = (__bf16)hnew;        // next layer's A operand
        }
        __syncthreads();             // h writes visible before next step reads
    }
}

// ---------------- kernel 7: final gather ----------------
__global__ void k_final(const float* __restrict__ hout, const int* __restrict__ lengths,
                        float* __restrict__ out) {
    int k = blockIdx.x;
    int h = threadIdx.x;
    int t = lengths[k] - 1;
    out[k * H + h] = hout[((size_t)k * NB + t) * H + h];
}

// ---------------- host launch ----------------
extern "C" void kernel_launch(void* const* d_in, const int* in_sizes, int n_in,
                              void* d_out, int out_size, void* d_ws, size_t ws_size,
                              hipStream_t stream) {
    const float* fbp   = (const float*)d_in[0];   // [K, B]
    const int*   seq   = (const int*)  d_in[1];   // [K, B]
    // d_in[2] = uid (unused by reference)
    const float* embed = (const float*)d_in[3];   // [NP+1, H]
    const float* wih   = (const float*)d_in[4];   // [L, 3H, H]
    const float* whh   = (const float*)d_in[5];   // [L, 3H, H]
    const float* bih   = (const float*)d_in[6];   // [L, 3H]
    const float* bhh   = (const float*)d_in[7];   // [L, 3H]
    float* out = (float*)d_out;                   // [K, H]

    // workspace carve-up (256B aligned)
    char* ws = (char*)d_ws;
    size_t off = 0;
    auto carve = [&](size_t bytes) { char* p = ws + off; off = (off + bytes + 255) & ~(size_t)255; return p; };
    __bf16* wih_pk = (__bf16*)carve(L * WPL * 2);
    __bf16* whh_pk = (__bf16*)carve(L * WPL * 2);
    float*  fmax   = (float*) carve((size_t)K_USERS * NB * 4);
    float*  bke    = (float*) carve((size_t)K_USERS * NB * H * 4);
    int*    order  = (int*)   carve((size_t)K_USERS * NB * 4);
    int*    lens   = (int*)   carve((size_t)K_USERS * 4);
    __bf16* xbf    = (__bf16*)carve((size_t)K_USERS * NB * H * 2);
    float*  gibuf  = (float*) carve((size_t)K_USERS * NB * H3 * 4);
    float*  hout   = (float*) carve((size_t)K_USERS * NB * H * 4);

    const int packN = (int)(L * WPL);             // 393216 elements each
    // 1) pack weights into WMMA fragment order (f32 -> bf16)
    k_pack<<<packN / 256, 256, 0, stream>>>(wih, wih_pk);
    k_pack<<<packN / 256, 256, 0, stream>>>(whh, whh_pk);
    // 2) basket pooling
    k_basket<<<K_USERS * NB, H, 0, stream>>>(fbp, seq, embed, bke, fmax);
    // 3) order + lengths
    k_order<<<1, K_USERS, 0, stream>>>(fmax, order, lens);
    // 4) reorder -> bf16 x0
    k_reorder<<<(K_USERS * NB * H) / 256, 256, 0, stream>>>(bke, order, xbf);
    // 5) two GRU layers
    for (int l = 0; l < L; ++l) {
        const __bf16* wihl = wih_pk + (size_t)l * WPL;
        const __bf16* whhl = whh_pk + (size_t)l * WPL;
        const float*  bihl = bih + l * H3;
        const float*  bhhl = bhh + l * H3;
        k_gi<<<dim3(MTOT / 32, H3 / 256), 256, 0, stream>>>(xbf, wihl, bihl, gibuf);
        k_scan<<<K_USERS / 16, 512, 0, stream>>>(gibuf, whhl, bhhl, hout, xbf);
    }
    // 6) gather h[lengths-1]
    k_final<<<K_USERS, H, 0, stream>>>(hout, lens, out);
}
